// ChiralEmbeddingModel_12275016532149
// MI455X (gfx1250) — compile-verified, matrix operand order checked by reading, not run on hardware
//
#include <hip/hip_runtime.h>
#include <hip/hip_bf16.h>

// ---------------------------------------------------------------------------
// ChiralEmbeddingModel forward for MI455X (gfx1250, wave32, WMMA).
//
// bf16 inputs + f32 accumulation via v_wmma_f32_16x16x32_bf16.
//   - block = 128 threads (4 waves), 64 atoms per block, grid = 2048
//   - block prefetches its whole 96KB x-slab (global_prefetch_b8) before the
//     weight-staging phase so phase-1 loads hit cache (1 wave/SIMD otherwise
//     has little latency hiding).
//   - weights staged once per block into LDS as bf16, PRE-SWIZZLED into WMMA
//     B-fragment order: a fragment load is 2x ds_load_b128 per lane.
//   - A operands row-major in per-wave LDS -> fragment = 2x ds_load_b128.
//   - xn kept in TWO layouts: row-major (A fragments; hoisted to registers,
//     then the buffer is reused for `cross`) and transposed xnT[j][u][atom]
//     (elementwise stages read 8 rows per lane as ONE ds_load_b128).
//   - elementwise stores use per-lane base pointers + constant offsets and a
//     1-add bf16 conversion (d16_hi-friendly); VALU co-executes with WMMA.
// ---------------------------------------------------------------------------

#define N_ATOMS 131072
#define INV     256
#define MUL     128
#define OUT     512
#define ROWLEN  640           // INV + MUL*3
#define EPSF    1e-6f

// C1 = sqrt(3/256), C2 = sqrt(1/384)
#define C1F 0.10825317547305482f
#define C2F 0.05103103630798288f

typedef __attribute__((ext_vector_type(16))) __bf16 v16bf;
typedef __attribute__((ext_vector_type(8)))  float  v8f;

// staging-time conversion: full round-to-nearest-even (one-time cost)
__device__ __forceinline__ unsigned short f32_to_bf16_rne(float f) {
    unsigned int x = __float_as_uint(f);
    unsigned int r = x + 0x7FFFu + ((x >> 16) & 1u);
    return (unsigned short)(r >> 16);
}
// hot-path conversion: round-half-up = one v_add, value lands in bits[31:16]
__device__ __forceinline__ unsigned short f32_to_bf16_rhu(float f) {
    return (unsigned short)((__float_as_uint(f) + 0x8000u) >> 16);
}
__device__ __forceinline__ float bf16u_to_f32(unsigned short u) {
    return __uint_as_float(((unsigned int)u) << 16);
}

union FragU  { v16bf v; uint4 q[2]; };
union ColU   { uint4 q; unsigned short h[8]; };

// A fragment (16x32 bf16) from row-major [16][ldk] LDS, K window at kbase.
// ISA 16-bit A layout: each lane reads two contiguous 16B runs of K.
__device__ __forceinline__ v16bf load_frag_a_rm(const unsigned short* __restrict__ src,
                                                int ldk, int kbase, int lane) {
    const int half = lane >> 4;
    const int m    = lane & 15;
    const unsigned short* p = src + m * ldk + kbase + half * 8;
    FragU u;
    u.q[0] = *(const uint4*)(p);        // K = kbase + half*8 + [0..7]
    u.q[1] = *(const uint4*)(p + 16);   // K = kbase + 16 + half*8 + [0..7]
    return u.v;
}

// B fragment (32x16 bf16) from pre-swizzled fragment storage:
// 32 contiguous bytes per lane -> 2x ds_load_b128.
__device__ __forceinline__ v16bf load_frag_b_sw(const unsigned short* __restrict__ base,
                                                int tile, int lane) {
    const unsigned short* p = base + (tile * 32 + lane) * 16;
    FragU u;
    u.q[0] = *(const uint4*)(p);
    u.q[1] = *(const uint4*)(p + 8);
    return u.v;
}

__global__ __launch_bounds__(128, 1)
void chiral_embed_fwd(const float* __restrict__ emb,
                      const float* __restrict__ rms_g,
                      const float* __restrict__ w1,
                      const float* __restrict__ w2,
                      const float* __restrict__ W_out,
                      const float* __restrict__ b_out,
                      float* __restrict__ out) {
    extern __shared__ char smem_raw[];
    unsigned short* s_w1F = (unsigned short*)smem_raw;           // 8nt*4kc frag tiles
    unsigned short* s_w2F = s_w1F + MUL * MUL;
    unsigned short* s_WoF = s_w2F + MUL * MUL;                   // 32nt*4kc frag tiles
    float*          s_rmsg = (float*)(s_WoF + MUL * OUT);        // [128]
    unsigned short* s_wavebase = (unsigned short*)(s_rmsg + MUL);

    const int tid  = threadIdx.x;
    const int lane = tid & 31;
    const int wave = tid >> 5;

    // per-wave: s_xc = xn row-major [3][16][128] (reused for cross after the
    // phase-2 A fragments are hoisted), s_xnT = xn transposed [3][128][16],
    // s_chi = chi row-major [16][128]
    unsigned short* s_xc  = s_wavebase + wave * (3*16*MUL + 3*MUL*16 + 16*MUL);
    unsigned short* s_xnT = s_xc + 3 * 16 * MUL;
    unsigned short* s_chi = s_xnT + 3 * MUL * 16;

    // ---- prefetch this block's x-slab while we stage weights --------------
    // 64 atoms x 384 floats = 96KB; 12 cachelines (128B) per atom.
    {
        const long blk_atom0 = (long)blockIdx.x * 64;
        for (int i = tid; i < 64 * 12; i += (int)blockDim.x) {
            const int a = i / 12, line = i - a * 12;
            __builtin_prefetch(emb + (blk_atom0 + a) * ROWLEN + INV + line * 32, 0, 0);
        }
    }

    // ---- stage weights: bf16, pre-swizzled into B-fragment order ----------
    // element e of lane l in fragment (nt,kc):
    //   k = kc*32 + (e>=8)*16 + (l>=16)*8 + (e&7);  n = nt*16 + (l&15)
    // B[k][n] = W[n][k]  (B = W^T for all three GEMMs)
    for (int i = tid; i < MUL * MUL; i += blockDim.x) {
        const int e  = i & 15;
        const int l  = (i >> 4) & 31;
        const int kc = (i >> 9) & 3;
        const int nt = (i >> 11) & 7;
        const int k  = kc * 32 + ((e >> 3) << 4) + ((l >> 4) << 3) + (e & 7);
        const int n  = nt * 16 + (l & 15);
        s_w1F[i] = f32_to_bf16_rne(w1[n * MUL + k]);
        s_w2F[i] = f32_to_bf16_rne(w2[n * MUL + k]);
    }
    for (int i = tid; i < OUT * MUL; i += blockDim.x) {
        const int e  = i & 15;
        const int l  = (i >> 4) & 31;
        const int kc = (i >> 9) & 3;
        const int nt = (i >> 11) & 31;
        const int k  = kc * 32 + ((e >> 3) << 4) + ((l >> 4) << 3) + (e & 7);
        const int n  = nt * 16 + (l & 15);
        s_WoF[i] = f32_to_bf16_rne(W_out[n * MUL + k]);
    }
    if (tid < MUL) s_rmsg[tid] = rms_g[tid];
    __syncthreads();

    const long atom_base = (long)blockIdx.x * 64 + (long)wave * 16;

    // ---- phase 1: load x, RMS-normalize, write xn in both layouts ---------
    for (int a = 0; a < 16; ++a) {
        const float* px = emb + (atom_base + a) * ROWLEN + INV;
        float vals[12];
        float ss = 0.f;
#pragma unroll
        for (int it = 0; it < 12; ++it) {
            const float v = px[lane + 32 * it];
            vals[it] = v;
            ss += v * v;
        }
#pragma unroll
        for (int off = 16; off >= 1; off >>= 1)
            ss += __shfl_xor(ss, off, 32);
        const float scale = rsqrtf(ss * (1.0f / 384.0f) + EPSF);
#pragma unroll
        for (int it = 0; it < 12; ++it) {
            const int f = lane + 32 * it;
            const int u = f / 3, j = f - u * 3;
            const unsigned short b = f32_to_bf16_rne(vals[it] * s_rmsg[u] * scale);
            s_xc[(j * 16 + a) * MUL + u]  = b;      // row-major (for A fragments)
            s_xnT[(j * MUL + u) * 16 + a] = b;      // transposed (for elementwise)
        }
    }

    const int half8 = (lane >> 4) << 3;             // C/D layout row base
    const int ncol  = lane & 15;

    // per-lane base pointers; all further indices are compile-time constants
    unsigned short* pcw = s_xc  + half8 * MUL + ncol;        // cross writes
    const unsigned short* pxr = s_xnT + ncol * 16 + half8;   // xn column reads
    unsigned short* pkw = s_chi + half8 * MUL + ncol;        // chi writes

    // ---- phase 2: y = xn @ w1^T (per j), cross = C1 * (xn x y) ------------
    {
        v16bf axn[3][4];
#pragma unroll
        for (int j = 0; j < 3; ++j)
#pragma unroll
            for (int kc = 0; kc < 4; ++kc)
                axn[j][kc] = load_frag_a_rm(s_xc + j * 16 * MUL, MUL, kc * 32, lane);

        for (int t = 0; t < 8; ++t) {
            v8f acc[3];
#pragma unroll
            for (int j = 0; j < 3; ++j) {
                v8f c = {};
#pragma unroll
                for (int kc = 0; kc < 4; ++kc) {
                    v16bf bf = load_frag_b_sw(s_w1F, t * 4 + kc, lane);
                    c = __builtin_amdgcn_wmma_f32_16x16x32_bf16(
                            false, axn[j][kc], false, bf, (short)0, c, false, false);
                }
                acc[j] = c;
            }
            ColU xq[3];                              // xn[j][half8..half8+7][n]
#pragma unroll
            for (int j = 0; j < 3; ++j)
                xq[j].q = *(const uint4*)(pxr + t * 256 + j * 2048);
            unsigned short* pw = pcw + t * 16;
#pragma unroll
            for (int r = 0; r < 8; ++r) {
                const float x0 = bf16u_to_f32(xq[0].h[r]);
                const float x1 = bf16u_to_f32(xq[1].h[r]);
                const float x2 = bf16u_to_f32(xq[2].h[r]);
                const float y0 = acc[0][r], y1 = acc[1][r], y2 = acc[2][r];
                // cross overwrites the (now dead) row-major xn buffer
                pw[r * MUL + 0 * 16 * MUL] = f32_to_bf16_rhu(C1F * (x1 * y2 - x2 * y1));
                pw[r * MUL + 1 * 16 * MUL] = f32_to_bf16_rhu(C1F * (x2 * y0 - x0 * y2));
                pw[r * MUL + 2 * 16 * MUL] = f32_to_bf16_rhu(C1F * (x0 * y1 - x1 * y0));
            }
        }
    }

    // ---- phase 3: z = cross @ w2^T, chi = C2 * dot(xn, z) -----------------
    {
        v16bf acr[3][4];
#pragma unroll
        for (int j = 0; j < 3; ++j)
#pragma unroll
            for (int kc = 0; kc < 4; ++kc)
                acr[j][kc] = load_frag_a_rm(s_xc + j * 16 * MUL, MUL, kc * 32, lane);

        for (int t = 0; t < 8; ++t) {
            v8f acc[3];
#pragma unroll
            for (int j = 0; j < 3; ++j) {
                v8f c = {};
#pragma unroll
                for (int kc = 0; kc < 4; ++kc) {
                    v16bf bf = load_frag_b_sw(s_w2F, t * 4 + kc, lane);
                    c = __builtin_amdgcn_wmma_f32_16x16x32_bf16(
                            false, acr[j][kc], false, bf, (short)0, c, false, false);
                }
                acc[j] = c;
            }
            ColU xq[3];
#pragma unroll
            for (int j = 0; j < 3; ++j)
                xq[j].q = *(const uint4*)(pxr + t * 256 + j * 2048);
            unsigned short* pw = pkw + t * 16;
#pragma unroll
            for (int r = 0; r < 8; ++r) {
                const float x0 = bf16u_to_f32(xq[0].h[r]);
                const float x1 = bf16u_to_f32(xq[1].h[r]);
                const float x2 = bf16u_to_f32(xq[2].h[r]);
                const float chi = C2F * (x0 * acc[0][r] + x1 * acc[1][r] + x2 * acc[2][r]);
                pw[r * MUL] = f32_to_bf16_rhu(chi);
            }
        }
    }

    // ---- phase 4: out = chi @ W_out^T + b_out -----------------------------
    {
        v16bf achi[4];
#pragma unroll
        for (int kc = 0; kc < 4; ++kc)
            achi[kc] = load_frag_a_rm(s_chi, MUL, kc * 32, lane);

        float* po = out + (atom_base + half8) * OUT + ncol;  // + t*16 per tile
        for (int t = 0; t < 32; ++t) {
            v8f c = {};
#pragma unroll
            for (int kc = 0; kc < 4; ++kc) {
                v16bf bf = load_frag_b_sw(s_WoF, t * 4 + kc, lane);
                c = __builtin_amdgcn_wmma_f32_16x16x32_bf16(
                        false, achi[kc], false, bf, (short)0, c, false, false);
            }
            const float bias = b_out[t * 16 + ncol];
            float* pw = po + t * 16;
#pragma unroll
            for (int r = 0; r < 8; ++r)
                pw[r * OUT] = c[r] + bias;
        }
    }
}

extern "C" void kernel_launch(void* const* d_in, const int* in_sizes, int n_in,
                              void* d_out, int out_size, void* d_ws, size_t ws_size,
                              hipStream_t stream) {
    (void)in_sizes; (void)n_in; (void)out_size; (void)d_ws; (void)ws_size;
    const float* emb   = (const float*)d_in[0];   // [N, 640]
    const float* rmsg  = (const float*)d_in[1];   // [128]
    const float* w1    = (const float*)d_in[2];   // [128,128]
    const float* w2    = (const float*)d_in[3];   // [128,128]
    const float* W_out = (const float*)d_in[4];   // [512,128]
    const float* b_out = (const float*)d_in[5];   // [512]
    float* out = (float*)d_out;                   // [N, 512]

    // LDS: w1F + w2F (2*32KB) + WoF (128KB) + rms_g (512B)
    //      + 4 waves * (xc + xnT + chi) = 4 * 28KB  => 311,808 bytes
    const size_t lds_bytes =
        (size_t)(MUL * MUL * 2 + MUL * OUT + 4 * (3*16*MUL + 3*MUL*16 + 16*MUL))
            * sizeof(unsigned short)
        + MUL * sizeof(float);

    static bool attr_set = false;   // idempotent host-side attribute
    if (!attr_set) {
        hipFuncSetAttribute((const void*)chiral_embed_fwd,
                            hipFuncAttributeMaxDynamicSharedMemorySize,
                            (int)lds_bytes);
        attr_set = true;
    }

    const int blocks = N_ATOMS / 64;   // 64 atoms per block (4 waves x 16)
    chiral_embed_fwd<<<blocks, 128, lds_bytes, stream>>>(
        emb, rmsg, w1, w2, W_out, b_out, out);
}
